// StructureLevelFeatureExtractor_23502061043726
// MI455X (gfx1250) — compile-verified, matrix operand order checked by reading, not run on hardware
//
#include <hip/hip_runtime.h>

#define HD 8      // heads
#define DD 32     // dim per head
#define F  256    // HD*DD
#define WSTR 40   // padded LDS stride (halves): 80B rows, 16B-aligned, bank-spread

typedef _Float16 v16h __attribute__((ext_vector_type(16)));
typedef float    v8f  __attribute__((ext_vector_type(8)));

// ---------- order-preserving float <-> uint encoding for atomicMax ----------
__device__ __forceinline__ unsigned fenc(float f) {
  unsigned u = __float_as_uint(f);
  return (u & 0x80000000u) ? ~u : (u | 0x80000000u);
}
__device__ __forceinline__ float fdec(unsigned u) {
  unsigned v = (u & 0x80000000u) ? (u & 0x7FFFFFFFu) : ~u;
  return __uint_as_float(v);
}
#define ENC_NEG_INF 0x007FFFFFu   // fenc(-inf)

// =====================================================================
// GEMM: Y[N,256] = X[N,K] @ W[K,256]   (K % 32 == 0)
// Wave = 16 rows x 256 cols (16 WMMA tiles, A reused, 128 acc VGPRs).
// W staged f32->f16 transposed into double-buffered LDS (ping-pong so
// global staging of tile t+1 overlaps the 16 WMMAs of tile t); B
// fragments software-pipelined so ds waits overlap XDL execution.
// =====================================================================
__global__ __launch_bounds__(256) void gat_gemm_wmma(
    const float* __restrict__ X, const float* __restrict__ W,
    float* __restrict__ Y, int N, int K)
{
  __shared__ _Float16 wl[2][F * WSTR];     // 2 x 20 KB ping-pong
  const int lane = threadIdx.x & 31;
  const int wid  = threadIdx.x >> 5;
  const int wavesPerBlk = blockDim.x >> 5;
  const int tile = blockIdx.x * wavesPerBlk + wid;
  const int r0   = tile * 16;
  const bool active = (r0 < N);
  const int m  = lane & 15;                // row (A) / col (B) within tile
  const int hi = lane >> 4;                // K-half selector

  v8f acc[16] = {};

  const int T = K >> 5;                    // number of 32-wide k tiles

  // stage k-tile 0
  for (int idx = threadIdx.x; idx < F * 32; idx += blockDim.x) {
    int kk = idx >> 8, n = idx & 255;
    wl[0][n * WSTR + kk] = (_Float16)W[(size_t)kk * F + n];
  }

  for (int kt = 0; kt < T; ++kt) {
    __syncthreads();                       // wl[kt&1] ready
    if (kt + 1 < T) {                      // stage next tile into other buffer
      int k0n = (kt + 1) << 5;
      _Float16* dstb = wl[(kt + 1) & 1];
      for (int idx = threadIdx.x; idx < F * 32; idx += blockDim.x) {
        int kk = idx >> 8, n = idx & 255;
        dstb[n * WSTR + kk] = (_Float16)W[(size_t)(k0n + kk) * F + n];
      }
    }
    if (active) {
      const _Float16* wb = wl[kt & 1];
      // A fragment: element i holds K = kt*32 + (i&7) + ((i>>3)<<4) + hi*8
      int arow = r0 + m; if (arow >= N) arow = N - 1;
      const float* xr = X + (size_t)arow * K + (kt << 5) + hi * 8;
      v16h a;
#pragma unroll
      for (int i = 0; i < 16; ++i) a[i] = (_Float16)xr[(i & 7) + ((i >> 3) << 4)];

      // software-pipelined B fragments
      v16h bcur;
      {
        const _Float16* wr = &wb[m * WSTR + hi * 8];
#pragma unroll
        for (int i = 0; i < 16; ++i) bcur[i] = wr[(i & 7) + ((i >> 3) << 4)];
      }
#pragma unroll
      for (int nt = 0; nt < 16; ++nt) {
        v16h bnext = bcur;
        if (nt < 15) {
          const _Float16* wr = &wb[((nt + 1) * 16 + m) * WSTR + hi * 8];
#pragma unroll
          for (int i = 0; i < 16; ++i) bnext[i] = wr[(i & 7) + ((i >> 3) << 4)];
        }
        acc[nt] = __builtin_amdgcn_wmma_f32_16x16x32_f16(
            false, a, false, bcur, (short)0, acc[nt], false, false);
        bcur = bnext;
      }
    }
  }
  if (!active) return;

  const int rowOff = hi * 8;               // C/D layout: vgpr r -> row r (+8 hi half)
#pragma unroll
  for (int nt = 0; nt < 16; ++nt) {
#pragma unroll
    for (int r = 0; r < 8; ++r) {
      int row = r0 + rowOff + r;
      if (row < N) Y[(size_t)row * F + nt * 16 + m] = acc[nt][r];
    }
  }
}

// ---------- per-(node,head) attention scores ----------
__global__ void gat_scores(const float* __restrict__ xw,
                           const float* __restrict__ a_s,
                           const float* __restrict__ a_d,
                           float* __restrict__ sc_s, float* __restrict__ sc_d, int N)
{
  int t = blockIdx.x * blockDim.x + threadIdx.x;
  if (t >= N * HD) return;
  int h = t & (HD - 1);
  const float* row = xw + (size_t)(t >> 3) * F + h * DD;
  float ss = 0.f, sd = 0.f;
#pragma unroll
  for (int d = 0; d < DD; ++d) {
    float v = row[d];
    ss += v * a_s[h * DD + d];
    sd += v * a_d[h * DD + d];
  }
  sc_s[t] = ss; sc_d[t] = sd;
}

// ---------- per-layer state init ----------
__global__ void layer_init(float* __restrict__ agg, unsigned* __restrict__ menc,
                           float* __restrict__ ssum, int N)
{
  int i = blockIdx.x * blockDim.x + threadIdx.x;
  if (i < N * F) agg[i] = 0.f;
  if (i < N * HD) { menc[i] = ENC_NEG_INF; ssum[i] = 0.f; }
}

// ---------- edge pass 1: alpha = leaky_relu(s[src]+d[dst]); segment max ----------
__global__ void edge_alpha_max(const int* __restrict__ ei, int E, int Etot,
                               const float* __restrict__ sc_s, const float* __restrict__ sc_d,
                               float* __restrict__ alpha, unsigned* __restrict__ menc)
{
  int idx = blockIdx.x * blockDim.x + threadIdx.x;
  if (idx >= Etot * HD) return;
  int e = idx >> 3, h = idx & (HD - 1);
  int s, d;
  if (e < E) { s = ei[e]; d = ei[e + E]; } else { s = d = e - E; }
  float a = sc_s[s * HD + h] + sc_d[d * HD + h];
  a = a > 0.f ? a : 0.2f * a;
  alpha[idx] = a;
  atomicMax(&menc[d * HD + h], fenc(a));
}

// ---------- edge pass 2: e = exp(alpha - max); segment sum ----------
__global__ void edge_exp_sum(const int* __restrict__ ei, int E, int Etot,
                             float* __restrict__ alpha, const unsigned* __restrict__ menc,
                             float* __restrict__ ssum)
{
  int idx = blockIdx.x * blockDim.x + threadIdx.x;
  if (idx >= Etot * HD) return;
  int e = idx >> 3, h = idx & (HD - 1);
  int d = (e < E) ? ei[e + E] : (e - E);
  float ev = __expf(alpha[idx] - fdec(menc[d * HD + h]));
  alpha[idx] = ev;
  atomicAdd(&ssum[d * HD + h], ev);
}

// ---------- edge pass 3: agg[dst] += w * xw[src] ----------
// Wave per edge: per-head softmax weights in lanes, broadcast via __shfl
// (lane permute, no LDS/barrier); 8 gathered reads + 8 L2 atomics per lane.
__global__ void edge_aggregate(const int* __restrict__ ei, int E, int Etot,
                               const float* __restrict__ eval, const float* __restrict__ ssum,
                               const float* __restrict__ xw, float* __restrict__ agg)
{
  int e = blockIdx.x * 8 + (threadIdx.x >> 5);
  if (e >= Etot) return;
  int lane = threadIdx.x & 31;
  int s, d;
  if (e < E) { s = ei[e]; d = ei[e + E]; } else { s = d = e - E; }
  int h = lane & (HD - 1);
  float wv = eval[(size_t)e * HD + h] / ssum[d * HD + h];
  const float* xr = xw + (size_t)s * F;
  float* ar = agg + (size_t)d * F;
#pragma unroll
  for (int i = 0; i < HD; ++i) {
    float w = __shfl(wv, i, 32);
    int f = i * 32 + lane;
    atomicAdd(&ar[f], xr[f] * w);
  }
}

// ---------- bias + relu -> next layer input ----------
__global__ void bias_relu(const float* __restrict__ agg, const float* __restrict__ b,
                          float* __restrict__ x, int total)
{
  int i = blockIdx.x * blockDim.x + threadIdx.x;
  if (i >= total) return;
  x[i] = fmaxf(agg[i] + b[i & (F - 1)], 0.f);
}

// ---------- misc ----------
__global__ void zero_vec(float* p) { p[threadIdx.x] = 0.f; }

__global__ void pool_partial(const float* __restrict__ x, float* __restrict__ g, int N)
{
  int f = threadIdx.x;
  float s = 0.f;
  for (int n = blockIdx.x; n < N; n += gridDim.x) s += x[(size_t)n * F + f];
  atomicAdd(&g[f], s);
}

// ---------- MHA(seq=1) == V path, then output proj, then fusion ----------
__global__ void head_fuse(const float* __restrict__ gc, const float* __restrict__ gd,
                          const float* __restrict__ in_w, const float* __restrict__ in_b,
                          const float* __restrict__ out_w, const float* __restrict__ out_b,
                          const float* __restrict__ fus_w, const float* __restrict__ fus_b,
                          float* __restrict__ out, float invNc, float invNd)
{
  __shared__ float pc[F], pd[F], vc[F], vd[F], fc[F], fd[F];
  int t = threadIdx.x;
  pc[t] = gc[t] * invNc; pd[t] = gd[t] * invNd;
  __syncthreads();
  // V projection: rows 512..767 of in_w (softmax over length-1 axis == 1)
  float sc = in_b[2 * F + t], sd = sc;
  const float* wr = in_w + (size_t)(2 * F + t) * F;
  for (int f = 0; f < F; ++f) { sc += pc[f] * wr[f]; sd += pd[f] * wr[f]; }
  vc[t] = sc; vd[t] = sd;
  __syncthreads();
  sc = out_b[t]; sd = out_b[t];
  const float* owr = out_w + (size_t)t * F;
  for (int j = 0; j < F; ++j) { sc += vc[j] * owr[j]; sd += vd[j] * owr[j]; }
  fc[t] = sc; fd[t] = sd;
  __syncthreads();
  float o = fus_b[t];
  const float* fwr = fus_w + (size_t)t * (2 * F);
  for (int q = 0; q < F; ++q) o += fc[q] * fwr[q] + fd[q] * fwr[F + q];
  out[t] = o;
}

extern "C" void kernel_launch(void* const* d_in, const int* in_sizes, int n_in,
                              void* d_out, int out_size, void* d_ws, size_t ws_size,
                              hipStream_t stream)
{
  (void)n_in; (void)out_size; (void)ws_size;
  const float* x0s[2] = { (const float*)d_in[0], (const float*)d_in[1] };
  const int*   eis[2] = { (const int*)d_in[2],   (const int*)d_in[3]   };
  const int K0g[2] = { in_sizes[4] / F, in_sizes[16] / F };          // 64, 32
  const int Ng[2]  = { in_sizes[0] / K0g[0], in_sizes[1] / K0g[1] }; // 20000
  const int Eg[2]  = { in_sizes[2] / 2, in_sizes[3] / 2 };           // 320000
  const int Nmax = Ng[0] > Ng[1] ? Ng[0] : Ng[1];
  const int Et0 = Eg[0] + Ng[0], Et1 = Eg[1] + Ng[1];
  const int EtotMax = Et0 > Et1 ? Et0 : Et1;

  size_t off = 0;
  auto alloc = [&](size_t bytes) -> void* {
    void* p = (char*)d_ws + off;
    off += (bytes + 255) & ~(size_t)255;
    return p;
  };
  float*    X1   = (float*)alloc((size_t)Nmax * F * 4);
  float*    XW   = (float*)alloc((size_t)Nmax * F * 4);
  float*    AGG  = (float*)alloc((size_t)Nmax * F * 4);
  float*    EV   = (float*)alloc((size_t)EtotMax * HD * 4);
  float*    SCs  = (float*)alloc((size_t)Nmax * HD * 4);
  float*    SCd  = (float*)alloc((size_t)Nmax * HD * 4);
  unsigned* ME   = (unsigned*)alloc((size_t)Nmax * HD * 4);
  float*    SSum = (float*)alloc((size_t)Nmax * HD * 4);
  float*    G    = (float*)alloc(2 * F * 4);

  zero_vec<<<1, 2 * F, 0, stream>>>(G);

  for (int g = 0; g < 2; ++g) {
    const int N = Ng[g], E = Eg[g], Etot = E + N;
    const int* ei = eis[g];
    const int pbase = g ? 16 : 4;
    const float* xin = x0s[g];
    int K = K0g[g];
    for (int layer = 0; layer < 3; ++layer) {
      const float* lin = (const float*)d_in[pbase + 4 * layer + 0];
      const float* as_ = (const float*)d_in[pbase + 4 * layer + 1];
      const float* ad_ = (const float*)d_in[pbase + 4 * layer + 2];
      const float* bb  = (const float*)d_in[pbase + 4 * layer + 3];

      int tiles = (N + 15) / 16;
      gat_gemm_wmma<<<(tiles + 7) / 8, 256, 0, stream>>>(xin, lin, XW, N, K);
      gat_scores<<<(N * HD + 255) / 256, 256, 0, stream>>>(XW, as_, ad_, SCs, SCd, N);
      layer_init<<<(N * F + 255) / 256, 256, 0, stream>>>(AGG, ME, SSum, N);
      edge_alpha_max<<<(Etot * HD + 255) / 256, 256, 0, stream>>>(ei, E, Etot, SCs, SCd, EV, ME);
      edge_exp_sum<<<(Etot * HD + 255) / 256, 256, 0, stream>>>(ei, E, Etot, EV, ME, SSum);
      edge_aggregate<<<(Etot + 7) / 8, 256, 0, stream>>>(ei, E, Etot, EV, SSum, XW, AGG);
      bias_relu<<<(N * F + 255) / 256, 256, 0, stream>>>(AGG, bb, X1, N * F);

      xin = X1; K = F;
    }
    pool_partial<<<128, F, 0, stream>>>(X1, G + g * F, Ng[g]);
  }

  head_fuse<<<1, F, 0, stream>>>(G, G + F,
      (const float*)d_in[28], (const float*)d_in[29],
      (const float*)d_in[30], (const float*)d_in[31],
      (const float*)d_in[32], (const float*)d_in[33],
      (float*)d_out, 1.0f / (float)Ng[0], 1.0f / (float)Ng[1]);
}
